// MANNCell_90434831385056
// MI455X (gfx1250) — matrix operand order repaired and checked by reading, not compiled, not run on hardware
//
#include <hip/hip_runtime.h>
#include <hip/hip_bf16.h>
#include <math.h>

#define B_   128
#define H_   512
#define N_   2048
#define D_   256
#define R_   4
#define IN_  1536
#define G4H_ 2048            // 4*H
#define NP_  1028            // R*(D+1)
#define EPS_ 1e-8f
#define NWG_SCAN 16

typedef __attribute__((ext_vector_type(16))) __bf16 v16bf;
typedef __attribute__((ext_vector_type(8)))  float  v8f;

// ---------- bf16 helpers (round-to-nearest-even) ----------
__device__ __forceinline__ __bf16 f2bf(float f) {
  unsigned u = __builtin_bit_cast(unsigned, f);
  unsigned r = (u + 0x7FFFu + ((u >> 16) & 1u)) >> 16;
  unsigned short s = (unsigned short)r;
  return __builtin_bit_cast(__bf16, s);
}
__device__ __forceinline__ unsigned short f2bf_bits(float f) {
  unsigned u = __builtin_bit_cast(unsigned, f);
  return (unsigned short)((u + 0x7FFFu + ((u >> 16) & 1u)) >> 16);
}
__device__ __forceinline__ void fill_lo(v16bf& v, float4 a, float4 b) {
  v[0]=f2bf(a.x); v[1]=f2bf(a.y); v[2]=f2bf(a.z); v[3]=f2bf(a.w);
  v[4]=f2bf(b.x); v[5]=f2bf(b.y); v[6]=f2bf(b.z); v[7]=f2bf(b.w);
}
__device__ __forceinline__ void fill_hi(v16bf& v, float4 a, float4 b) {
  v[8]=f2bf(a.x);  v[9]=f2bf(a.y);  v[10]=f2bf(a.z); v[11]=f2bf(a.w);
  v[12]=f2bf(b.x); v[13]=f2bf(b.y); v[14]=f2bf(b.z); v[15]=f2bf(b.w);
}
__device__ __forceinline__ float sigmoidf_(float x){ return 1.0f/(1.0f+expf(-x)); }

// ---------- K0: ctrl_in concat:  X[b, c] = c<512 ? inputs : read_vectors ----------
__global__ void k0_concat(const float* __restrict__ inputs,
                          const float* __restrict__ rv,   // (R,B,D)
                          float* __restrict__ X) {
  int t = blockIdx.x * blockDim.x + threadIdx.x;
  if (t >= B_ * IN_) return;
  int b = t / IN_, c = t % IN_;
  float v;
  if (c < 512) v = inputs[b * 512 + c];
  else { int idx = c - 512; int r = idx >> 8, d = idx & 255;
         v = rv[((size_t)r * B_ + b) * D_ + d]; }
  X[t] = v;
}

// ---------- K_init: h_buf[0] = h0, zero barrier counters ----------
__global__ void k_init(const float* __restrict__ h0,
                       float* __restrict__ h_buf, int* __restrict__ bar) {
  int t = threadIdx.x;
  h_buf[t] = h0[t];
  h_buf[t + 256] = h0[t + 256];
  if (t == 0) { bar[0] = 0; bar[1] = 0; }
}

// ---------- K1: gates_x = X @ W_ih^T + (b_ih + b_hh)   [128 x 2048], bf16 WMMA ----------
__global__ void k1_gates_gemm(const float* __restrict__ X,
                              const float* __restrict__ W_ih,
                              const float* __restrict__ b_ih,
                              const float* __restrict__ b_hh,
                              float* __restrict__ gates) {
  int wave = (blockIdx.x * blockDim.x + threadIdx.x) >> 5;
  int lane = threadIdx.x & 31;
  int mt = wave >> 7, nt = wave & 127;           // 8 x 128 tiles
  int m0 = mt * 16, j0 = nt * 16;
  int half = lane >> 4, lm = lane & 15;
  int arow = m0 + lm;                            // batch row
  int jcol = j0 + lm;                            // output col
  v8f acc = {};
  for (int k0 = 0; k0 < IN_; k0 += 32) {
    const float4* pa0 = (const float4*)(X + (size_t)arow * IN_ + k0 + 8 * half);
    const float4* pa1 = (const float4*)(X + (size_t)arow * IN_ + k0 + 16 + 8 * half);
    v16bf av; fill_lo(av, pa0[0], pa0[1]); fill_hi(av, pa1[0], pa1[1]);
    const float4* pb = (const float4*)(W_ih + (size_t)jcol * IN_ + k0 + 8 * half);
    v16bf bv; fill_lo(bv, pb[0], pb[1]); fill_hi(bv, pb[4], pb[5]);
    acc = __builtin_amdgcn_wmma_f32_16x16x32_bf16(false, av, false, bv,
                                                  (short)0, acc, false, false);
  }
  float bias = b_ih[jcol] + b_hh[jcol];
#pragma unroll
  for (int v = 0; v < 8; ++v) {
    int brow = m0 + v + 8 * half;
    gates[(size_t)brow * G4H_ + jcol] = acc[v] + bias;
  }
}

// ---------- K2: sequential LSTM scan, 16 WGs, W_hh slice in LDS as packed bf16 ----------
__global__ void k2_lstm_scan(const float* __restrict__ gates,
                             const float* __restrict__ W_hh,
                             const float* __restrict__ c0,
                             float* __restrict__ h_buf,   // [2][512]
                             int* __restrict__ bar,       // [cnt, gen]
                             float* __restrict__ out) {   // d_out, stride 1536
  extern __shared__ char smem[];
  unsigned* w_lds = (unsigned*)smem;                  // [128][256] packed bf16 pairs
  float* h_sh = (float*)(smem + 131072);              // [512]
  float* g_sh = (float*)(smem + 131072 + 2048);       // [128]
  int tid = threadIdx.x;
  int u0 = blockIdx.x * 32;
  int row = tid >> 1, kh = tid & 1;                   // row 0..127, k-half
  int gate = row >> 5, unit = row & 31;
  int grow = gate * H_ + u0 + unit;                   // W_hh / gates_x row
  // preload W_hh slice -> bf16 pairs
  for (int i = 0; i < 128; ++i) {
    int k = kh * 256 + 2 * i;
    unsigned lo = f2bf_bits(W_hh[(size_t)grow * H_ + k]);
    unsigned hi = f2bf_bits(W_hh[(size_t)grow * H_ + k + 1]);
    w_lds[row * 256 + kh * 128 + i] = lo | (hi << 16);
  }
  float c = (tid < 32) ? c0[u0 + tid] : 0.0f;
  __syncthreads();

  for (int t = 0; t < B_; ++t) {
    int cur = t & 1;
    h_sh[tid * 2]     = __hip_atomic_load(&h_buf[cur * 512 + tid * 2],
                          __ATOMIC_ACQUIRE, __HIP_MEMORY_SCOPE_AGENT);
    h_sh[tid * 2 + 1] = __hip_atomic_load(&h_buf[cur * 512 + tid * 2 + 1],
                          __ATOMIC_ACQUIRE, __HIP_MEMORY_SCOPE_AGENT);
    __syncthreads();
    float acc = 0.0f;
    const float2* hp = (const float2*)(h_sh + kh * 256);
    const unsigned* wp = w_lds + row * 256 + kh * 128;
#pragma unroll 8
    for (int i = 0; i < 128; ++i) {
      unsigned w = wp[i];
      float w0 = __builtin_bit_cast(float, w << 16);
      float w1 = __builtin_bit_cast(float, w & 0xFFFF0000u);
      float2 hh = hp[i];
      acc += w0 * hh.x + w1 * hh.y;
    }
    acc += __shfl_xor(acc, 1);
    if (kh == 0) g_sh[row] = acc + gates[(size_t)t * G4H_ + grow];
    __syncthreads();
    if (tid < 32) {
      float gi = g_sh[tid], gf = g_sh[32 + tid], gg = g_sh[64 + tid], go = g_sh[96 + tid];
      c = sigmoidf_(gf) * c + sigmoidf_(gi) * tanhf(gg);
      float h = sigmoidf_(go) * tanhf(c);
      out[(size_t)t * (H_ + R_ * D_) + u0 + tid] = h;
      __hip_atomic_store(&h_buf[((t + 1) & 1) * 512 + u0 + tid], h,
                         __ATOMIC_RELEASE, __HIP_MEMORY_SCOPE_AGENT);
    }
    __syncthreads();
    if (tid == 0) {   // grid barrier across 16 WGs (sense via monotonically rising gen)
      int g = __hip_atomic_load(&bar[1], __ATOMIC_ACQUIRE, __HIP_MEMORY_SCOPE_AGENT);
      int a = __hip_atomic_fetch_add(&bar[0], 1, __ATOMIC_ACQ_REL, __HIP_MEMORY_SCOPE_AGENT);
      if (a == NWG_SCAN - 1) {
        __hip_atomic_store(&bar[0], 0, __ATOMIC_RELAXED, __HIP_MEMORY_SCOPE_AGENT);
        __hip_atomic_fetch_add(&bar[1], 1, __ATOMIC_ACQ_REL, __HIP_MEMORY_SCOPE_AGENT);
      } else {
        while (__hip_atomic_load(&bar[1], __ATOMIC_ACQUIRE, __HIP_MEMORY_SCOPE_AGENT) == g)
          __builtin_amdgcn_s_sleep(1);
      }
    }
    __syncthreads();
  }
}

// ---------- K3: params = ctrl_out @ W_p^T + b_p ; k=tanh, alpha=sigmoid ----------
__global__ void k3_params_gemm(const float* __restrict__ ctrl,  // d_out, stride 1536
                               const float* __restrict__ W_p,
                               const float* __restrict__ b_p,
                               float* __restrict__ kbuf,        // [B,R,D]
                               float* __restrict__ alpha) {     // [B,R]
  int wave = (blockIdx.x * blockDim.x + threadIdx.x) >> 5;
  if (wave >= 8 * 65) return;
  int lane = threadIdx.x & 31;
  int mt = wave / 65, nt = wave % 65;
  int m0 = mt * 16, j0 = nt * 16;
  int half = lane >> 4, lm = lane & 15;
  int arow = m0 + lm;
  int jcol = j0 + lm;
  __bf16 z = f2bf(0.0f);
  v16bf bz = {z,z,z,z,z,z,z,z,z,z,z,z,z,z,z,z};
  v8f acc = {};
  for (int k0 = 0; k0 < H_; k0 += 32) {
    const float4* pa0 = (const float4*)(ctrl + (size_t)arow * (H_ + R_ * D_) + k0 + 8 * half);
    const float4* pa1 = (const float4*)(ctrl + (size_t)arow * (H_ + R_ * D_) + k0 + 16 + 8 * half);
    v16bf av; fill_lo(av, pa0[0], pa0[1]); fill_hi(av, pa1[0], pa1[1]);
    v16bf bv = bz;
    if (jcol < NP_) {
      const float4* pb = (const float4*)(W_p + (size_t)jcol * H_ + k0 + 8 * half);
      fill_lo(bv, pb[0], pb[1]); fill_hi(bv, pb[4], pb[5]);
    }
    acc = __builtin_amdgcn_wmma_f32_16x16x32_bf16(false, av, false, bv,
                                                  (short)0, acc, false, false);
  }
  if (jcol < NP_) {
    float bias = b_p[jcol];
    int rr = jcol / 257, q = jcol - rr * 257;
#pragma unroll
    for (int v = 0; v < 8; ++v) {
      int brow = m0 + v + 8 * half;
      float val = acc[v] + bias;
      if (q == 256) alpha[brow * R_ + rr] = sigmoidf_(val);
      else          kbuf[((size_t)brow * R_ + rr) * D_ + q] = tanhf(val);
    }
  }
}

// ---------- K3b: k_norm[b,r] = ||k[b,r,:]|| ----------
__global__ void k3b_knorm(const float* __restrict__ kbuf, float* __restrict__ knorm) {
  int wave = (blockIdx.x * blockDim.x + threadIdx.x) >> 5;   // 512 waves
  int lane = threadIdx.x & 31;
  const float* kr = kbuf + (size_t)wave * D_;
  float s = 0.0f;
#pragma unroll
  for (int i = 0; i < 8; ++i) { float v = kr[lane * 8 + i]; s += v * v; }
  for (int off = 16; off; off >>= 1) s += __shfl_xor(s, off);
  if (lane == 0) knorm[wave] = sqrtf(s);
}

// ---------- K4: cosine scores K[b,r,n], fused M_norm, streaming M_prev (bf16 WMMA) ----------
__global__ void k4_scores(const float* __restrict__ M,
                          const float* __restrict__ kbuf,
                          const float* __restrict__ knorm,
                          float* __restrict__ scores) {
  int wave = (blockIdx.x * blockDim.x + threadIdx.x) >> 5;   // 16384 waves
  int lane = threadIdx.x & 31;
  int b = wave >> 7, nt = wave & 127;
  int n0 = nt * 16;
  int half = lane >> 4, lm = lane & 15;
  const float* Mb = M + (size_t)b * N_ * D_;
  int arow = n0 + lm;
  __bf16 z = f2bf(0.0f);
  v16bf bz = {z,z,z,z,z,z,z,z,z,z,z,z,z,z,z,z};
  float sq = 0.0f;
  v8f acc = {};
  for (int d0 = 0; d0 < D_; d0 += 32) {
    const float4* pa0 = (const float4*)(Mb + (size_t)arow * D_ + d0 + 8 * half);
    const float4* pa1 = (const float4*)(Mb + (size_t)arow * D_ + d0 + 16 + 8 * half);
    float4 a0 = pa0[0], a1 = pa0[1], a2 = pa1[0], a3 = pa1[1];
    sq += a0.x*a0.x + a0.y*a0.y + a0.z*a0.z + a0.w*a0.w;
    sq += a1.x*a1.x + a1.y*a1.y + a1.z*a1.z + a1.w*a1.w;
    sq += a2.x*a2.x + a2.y*a2.y + a2.z*a2.z + a2.w*a2.w;
    sq += a3.x*a3.x + a3.y*a3.y + a3.z*a3.z + a3.w*a3.w;
    v16bf av; fill_lo(av, a0, a1); fill_hi(av, a2, a3);
    v16bf bv = bz;
    if (lm < R_) {
      const float4* pb = (const float4*)(kbuf + ((size_t)b * R_ + lm) * D_ + d0 + 8 * half);
      fill_lo(bv, pb[0], pb[1]); fill_hi(bv, pb[4], pb[5]);
    }
    acc = __builtin_amdgcn_wmma_f32_16x16x32_bf16(false, av, false, bv,
                                                  (short)0, acc, false, false);
  }
  sq += __shfl_xor(sq, 16);                        // full row sum-of-squares
  float kn = (lm < R_) ? knorm[b * R_ + lm] : 1.0f;
#pragma unroll
  for (int v = 0; v < 8; ++v) {
    int m = v + 8 * half;
    float nr = __shfl(sq, m);                      // M_norm^2 of row n0+m
    float val = acc[v] / (sqrtf(nr) * kn + EPS_);
    if (lm < R_) scores[((size_t)b * R_ + lm) * N_ + n0 + m] = val;
  }
}

// ---------- K5: softmax over N, in place -> w_r ----------
__global__ void k5_softmax(float* __restrict__ scores) {
  __shared__ float red[256];
  float* row = scores + (size_t)blockIdx.x * N_;
  int tid = threadIdx.x;
  float vals[8];
  float mx = -INFINITY;
#pragma unroll
  for (int i = 0; i < 8; ++i) { vals[i] = row[tid + i * 256]; mx = fmaxf(mx, vals[i]); }
  red[tid] = mx; __syncthreads();
  for (int s = 128; s; s >>= 1) { if (tid < s) red[tid] = fmaxf(red[tid], red[tid + s]); __syncthreads(); }
  mx = red[0]; __syncthreads();
  float sum = 0.0f;
#pragma unroll
  for (int i = 0; i < 8; ++i) { vals[i] = expf(vals[i] - mx); sum += vals[i]; }
  red[tid] = sum; __syncthreads();
  for (int s = 128; s; s >>= 1) { if (tid < s) red[tid] += red[tid + s]; __syncthreads(); }
  float inv = 1.0f / red[0];
#pragma unroll
  for (int i = 0; i < 8; ++i) row[tid + i * 256] = vals[i] * inv;
}

// ---------- K5b: bottom-4 of w_u (tie-break: larger index, matching stable argsort) ----------
__global__ void k5b_bottom4(const float* __restrict__ wu, int* __restrict__ lu) {
  __shared__ float rv[256]; __shared__ int ri[256]; __shared__ int sel[4];
  int b = blockIdx.x, tid = threadIdx.x;
  const float* row = wu + (size_t)b * N_;
  for (int round = 0; round < 4; ++round) {
    float bv = INFINITY; int bi = -1;
    for (int i = tid; i < N_; i += 256) {
      bool skip = false;
      for (int j = 0; j < round; ++j) if (sel[j] == i) skip = true;
      float v = row[i];
      if (!skip && (v < bv || (v == bv && i > bi))) { bv = v; bi = i; }
    }
    rv[tid] = bv; ri[tid] = bi; __syncthreads();
    for (int s = 128; s; s >>= 1) {
      if (tid < s) {
        float ov = rv[tid + s]; int oi = ri[tid + s];
        if (ov < rv[tid] || (ov == rv[tid] && oi > ri[tid])) { rv[tid] = ov; ri[tid] = oi; }
      }
      __syncthreads();
    }
    if (tid == 0) { sel[round] = ri[0]; lu[b * 4 + round] = ri[0]; }
    __syncthreads();
  }
}

// ---------- K6a: S[b,r,r'] = alpha[r']*<w_r[r], w_r_prev[r']> + (1-alpha[r'])*sum_lu w_r[r] ----------
__global__ void k6a_S(const float* __restrict__ wr,     // [B,R,N] (post softmax)
                      const float* __restrict__ wrp,    // (R,B,N)
                      const float* __restrict__ alpha,  // [B,R]
                      const int* __restrict__ lu,
                      float* __restrict__ S) {
  __shared__ float red[16 * 256];
  __shared__ float rsum[4];
  int b = blockIdx.x, tid = threadIdx.x;
  const float* wrb = wr + (size_t)b * R_ * N_;
  float acc[16];
#pragma unroll
  for (int p = 0; p < 16; ++p) acc[p] = 0.0f;
  for (int n = tid; n < N_; n += 256) {
    float a0 = wrb[0 * N_ + n], a1 = wrb[1 * N_ + n], a2 = wrb[2 * N_ + n], a3 = wrb[3 * N_ + n];
    float p0 = wrp[((size_t)0 * B_ + b) * N_ + n];
    float p1 = wrp[((size_t)1 * B_ + b) * N_ + n];
    float p2 = wrp[((size_t)2 * B_ + b) * N_ + n];
    float p3 = wrp[((size_t)3 * B_ + b) * N_ + n];
    acc[0] += a0*p0; acc[1] += a0*p1; acc[2]  += a0*p2; acc[3]  += a0*p3;
    acc[4] += a1*p0; acc[5] += a1*p1; acc[6]  += a1*p2; acc[7]  += a1*p3;
    acc[8] += a2*p0; acc[9] += a2*p1; acc[10] += a2*p2; acc[11] += a2*p3;
    acc[12]+= a3*p0; acc[13]+= a3*p1; acc[14] += a3*p2; acc[15] += a3*p3;
  }
#pragma unroll
  for (int p = 0; p < 16; ++p) red[p * 256 + tid] = acc[p];
  __syncthreads();
  for (int s = 128; s; s >>= 1) {
    if (tid < s) {
#pragma unroll
      for (int p = 0; p < 16; ++p) red[p * 256 + tid] += red[p * 256 + tid + s];
    }
    __syncthreads();
  }
  if (tid < 4) {
    float t = 0.0f;
    for (int j = 0; j < 4; ++j) t += wrb[(size_t)tid * N_ + lu[b * 4 + j]];
    rsum[tid] = t;
  }
  __syncthreads();
  if (tid < 16) {
    int r = tid >> 2, rp = tid & 3;
    float al = alpha[b * R_ + rp];
    S[b * 16 + tid] = al * red[tid * 256] + (1.0f - al) * rsum[r];
  }
}

// ---------- K6b: reads = (w_r .* erase) @ M_prev + S @ k  ->  out[:, 512:1536] ----------
__global__ void k6b_reads(const float* __restrict__ M,
                          const float* __restrict__ wr,
                          const float* __restrict__ kbuf,
                          const float* __restrict__ S,
                          const int* __restrict__ lu,
                          float* __restrict__ out) {
  __shared__ float wsh[4 * 64];
  int b = blockIdx.x, tid = threadIdx.x;          // tid = d (D_=256)
  const float* Mb = M + (size_t)b * N_ * D_;
  const float* wrb = wr + (size_t)b * R_ * N_;
  int lu0 = lu[b * 4 + 0];
  float acc0 = 0, acc1 = 0, acc2 = 0, acc3 = 0;
  for (int nc = 0; nc < N_; nc += 64) {
    __syncthreads();
    int r = tid >> 6, i = tid & 63;
    int n = nc + i;
    float w = wrb[(size_t)r * N_ + n];
    if (n == lu0) w = 0.0f;                       // erase row
    wsh[r * 64 + i] = w;
    __syncthreads();
#pragma unroll 8
    for (int i2 = 0; i2 < 64; ++i2) {
      float m = Mb[(size_t)(nc + i2) * D_ + tid];
      acc0 += wsh[i2]       * m;
      acc1 += wsh[64 + i2]  * m;
      acc2 += wsh[128 + i2] * m;
      acc3 += wsh[192 + i2] * m;
    }
  }
  float k0 = kbuf[((size_t)b * R_ + 0) * D_ + tid];
  float k1 = kbuf[((size_t)b * R_ + 1) * D_ + tid];
  float k2 = kbuf[((size_t)b * R_ + 2) * D_ + tid];
  float k3 = kbuf[((size_t)b * R_ + 3) * D_ + tid];
  const float* Sb = S + b * 16;
  size_t base = (size_t)b * (H_ + R_ * D_) + H_;
  out[base + 0 * D_ + tid] = acc0 + Sb[0]*k0  + Sb[1]*k1  + Sb[2]*k2  + Sb[3]*k3;
  out[base + 1 * D_ + tid] = acc1 + Sb[4]*k0  + Sb[5]*k1  + Sb[6]*k2  + Sb[7]*k3;
  out[base + 2 * D_ + tid] = acc2 + Sb[8]*k0  + Sb[9]*k1  + Sb[10]*k2 + Sb[11]*k3;
  out[base + 3 * D_ + tid] = acc3 + Sb[12]*k0 + Sb[13]*k1 + Sb[14]*k2 + Sb[15]*k3;
}

// ---------- workspace layout (float offsets) ----------
static const size_t OFF_X      = 0;                         // 196608
static const size_t OFF_GATES  = 196608;                    // 262144
static const size_t OFF_K      = 458752;                    // 131072
static const size_t OFF_ALPHA  = 589824;                    // 512
static const size_t OFF_KNORM  = 590336;                    // 512
static const size_t OFF_SCORES = 590848;                    // 1048576 (-> w_r in place)
static const size_t OFF_HBUF   = 1639424;                   // 1024
static const size_t OFF_S      = 1640448;                   // 2048
static const size_t OFF_BAR    = 1642496;                   // 2 ints
static const size_t OFF_LU     = 1642512;                   // 512 ints

extern "C" void kernel_launch(void* const* d_in, const int* in_sizes, int n_in,
                              void* d_out, int out_size, void* d_ws, size_t ws_size,
                              hipStream_t stream) {
  (void)in_sizes; (void)n_in; (void)out_size; (void)ws_size;
  const float* inputs = (const float*)d_in[0];
  const float* h0     = (const float*)d_in[1];
  const float* c0     = (const float*)d_in[2];
  const float* rv     = (const float*)d_in[3];
  const float* wrp    = (const float*)d_in[4];
  const float* wu     = (const float*)d_in[5];
  const float* M      = (const float*)d_in[6];
  const float* W_ih   = (const float*)d_in[7];
  const float* W_hh   = (const float*)d_in[8];
  const float* b_ih   = (const float*)d_in[9];
  const float* b_hh   = (const float*)d_in[10];
  const float* W_p    = (const float*)d_in[11];
  const float* b_p    = (const float*)d_in[12];
  float* out = (float*)d_out;
  float* ws  = (float*)d_ws;

  float* X      = ws + OFF_X;
  float* gates  = ws + OFF_GATES;
  float* kbuf   = ws + OFF_K;
  float* alpha  = ws + OFF_ALPHA;
  float* knorm  = ws + OFF_KNORM;
  float* scores = ws + OFF_SCORES;
  float* hbuf   = ws + OFF_HBUF;
  float* Sbuf   = ws + OFF_S;
  int*   bar    = (int*)(ws + OFF_BAR);
  int*   lu     = (int*)(ws + OFF_LU);

  k0_concat     <<<768, 256, 0, stream>>>(inputs, rv, X);
  k_init        <<<1, 256, 0, stream>>>(h0, hbuf, bar);
  k1_gates_gemm <<<128, 256, 0, stream>>>(X, W_ih, b_ih, b_hh, gates);
  k2_lstm_scan  <<<NWG_SCAN, 256, 131072 + 2048 + 512, stream>>>(gates, W_hh, c0, hbuf, bar, out);
  k3_params_gemm<<<65, 256, 0, stream>>>(out, W_p, b_p, kbuf, alpha);
  k3b_knorm     <<<64, 256, 0, stream>>>(kbuf, knorm);
  k4_scores     <<<2048, 256, 0, stream>>>(M, kbuf, knorm, scores);
  k5_softmax    <<<512, 256, 0, stream>>>(scores);
  k5b_bottom4   <<<128, 256, 0, stream>>>(wu, lu);
  k6a_S         <<<128, 256, 0, stream>>>(scores, wrp, alpha, lu, Sbuf);
  k6b_reads     <<<128, 256, 0, stream>>>(M, scores, kbuf, Sbuf, lu, out);
}
